// MultiQueryAttention_49263274885935
// MI455X (gfx1250) — matrix operand order repaired and checked by reading, not compiled, not run on hardware
//
#include <hip/hip_runtime.h>
#include <hip/hip_bf16.h>
#include <math.h>

typedef __attribute__((ext_vector_type(16))) __bf16 v16bf;
typedef __attribute__((ext_vector_type(8)))  __bf16 v8bf;
typedef __attribute__((ext_vector_type(8)))  float  v8f;

#define NB     8
#define SEQ    1000
#define DMODEL 1024
#define NHEAD  16
#define HDIM   64
#define NTOK   (NB * SEQ)
#define NTOKP  (NTOK + 64)      // padded token count for unclamped vT reads
#define NKK    (DMODEL / 32)    // k-blocks per GEMM

static __device__ inline v8f wmma_bf16(v16bf a, v16bf b, v8f c) {
  return __builtin_amdgcn_wmma_f32_16x16x32_bf16(false, a, false, b, (short)0, c, false, false);
}

static __device__ inline v16bf cat8(v8bf lo, v8bf hi) {
  return __builtin_shufflevector(lo, hi, 0,1,2,3,4,5,6,7,8,9,10,11,12,13,14,15);
}

// ---------------------------------------------------------------------------
// One-shot prep kernels (run per launch, negligible cost).
// ---------------------------------------------------------------------------

// RoPE tables: cos/sin of t * 10000^(-2p/64), t<SEQ, p<32.
__global__ __launch_bounds__(256) void rope_table_kernel(float* __restrict__ ct,
                                                         float* __restrict__ st) {
  int idx = blockIdx.x * 256 + threadIdx.x;
  if (idx >= SEQ * (HDIM / 2)) return;
  int t = idx / (HDIM / 2);
  int p = idx % (HDIM / 2);
  float invf = expf(-(float)(2 * p) * (9.210340371976184f / (float)HDIM));
  float th = (float)t * invf;
  ct[idx] = cosf(th);
  st[idx] = sinf(th);
}

// Pack a (DMODEL x N) fp32 weight matrix into bf16 B-fragment order:
// layout [nt][kk][lane][16]; element i of lane = K (kk*32 + (i>>3)*16 + (lane>>4)*8 + (i&7)),
// column = nt*16 + (lane&15).  One thread per packed element.
__global__ __launch_bounds__(256) void pack_w_kernel(const float* __restrict__ W,
                                                     __bf16* __restrict__ Wp, int N) {
  int idx = blockIdx.x * 256 + threadIdx.x;
  if (idx >= DMODEL * N) return;
  int i    = idx & 15;
  int lane = (idx >> 4) & 31;
  int frag = idx >> 9;
  int kk   = frag % NKK;
  int nt   = frag / NKK;
  int k = kk * 32 + ((i >> 3) << 4) + ((lane >> 4) << 3) + (i & 7);
  int n = nt * 16 + (lane & 15);
  Wp[idx] = (__bf16)W[(size_t)k * N + n];
}

// ---------------------------------------------------------------------------
// LDS A-fragment staging helper: write element (m, d) of a 16 x DMODEL tile
// into packed A order [kk][lane][16].
// ---------------------------------------------------------------------------
static __device__ inline int a_pack_index(int m, int d) {
  int kk = d >> 5;
  int kd = d & 31;
  int h  = (kd >> 3) & 1;                       // lane half
  int i  = (kd & 7) + ((kd >> 4) << 3);         // element
  return ((kk << 5) + m + (h << 4)) * 16 + i;
}

// ---------------------------------------------------------------------------
// Kernel 1: fused Q/K/V projection + bias + RoPE (Q,K).
// Q,K stored bf16 row-major; V stored transposed vT[dim][token] (padded).
// ---------------------------------------------------------------------------
__global__ __launch_bounds__(256) void qkv_rope_kernel(
    const float* __restrict__ x,
    const __bf16* __restrict__ Wqp, const float* __restrict__ bq,
    const __bf16* __restrict__ Wkp, const float* __restrict__ bk,
    const __bf16* __restrict__ Wvp, const float* __restrict__ bv,
    const float* __restrict__ ct, const float* __restrict__ st,
    __bf16* __restrict__ qo, __bf16* __restrict__ ko, __bf16* __restrict__ vT)
{
  __shared__ __align__(64) __bf16 xs[16 * DMODEL];   // packed A fragments
  const int tid = threadIdx.x;
  const int m0  = blockIdx.x * 16;

  for (int idx = tid; idx < 16 * DMODEL; idx += 256) {
    int m = idx >> 10, d = idx & (DMODEL - 1);
    xs[a_pack_index(m, d)] = (__bf16)x[(size_t)(m0 + m) * DMODEL + d];
  }
  __syncthreads();

  const int wave = tid >> 5;
  const int lane = tid & 31;

  for (int tsel = wave * 9; tsel < wave * 9 + 9; ++tsel) {
    const __bf16* Wp; const float* bias; int nt, doRope, isQ, isK;
    if (tsel < 64)      { Wp = Wqp; bias = bq; nt = tsel;      doRope = 1; isQ = 1; isK = 0; }
    else if (tsel < 68) { Wp = Wkp; bias = bk; nt = tsel - 64; doRope = 1; isQ = 0; isK = 1; }
    else                { Wp = Wvp; bias = bv; nt = tsel - 68; doRope = 0; isQ = 0; isK = 0; }

    const __bf16* wbase = Wp + ((size_t)nt * NKK * 32 + lane) * 16;
    v8f acc = {};
    for (int kk = 0; kk < NKK; ++kk) {
      v16bf a  = *(const v16bf*)(xs + ((kk << 5) + lane) * 16);
      v16bf bf = *(const v16bf*)(wbase + (size_t)kk * 512);
      acc = wmma_bf16(a, bf, acc);
    }

    const int nl = lane & 15;
    const int c  = nt * 16 + nl;
    const float bb = bias[c];
#pragma unroll
    for (int i = 0; i < 8; ++i) {
      int mm  = i + ((lane >> 4) << 3);
      int row = m0 + mm;
      float val = acc[i] + bb;
      if (doRope) {
        int t    = row % SEQ;
        int pair = (c & (HDIM - 1)) >> 1;
        float cth = ct[t * (HDIM / 2) + pair];
        float sth = st[t * (HDIM / 2) + pair];
        float partner = __shfl_xor(val, 1, 32);     // even<->odd column
        val = val * cth + ((c & 1) ? partner * sth : -partner * sth);
      }
      if (isQ)      qo[(size_t)row * DMODEL + c] = (__bf16)val;
      else if (isK) ko[(size_t)row * HDIM   + c] = (__bf16)val;
      else          vT[(size_t)c * NTOKP + row]  = (__bf16)val;   // transposed V
    }
  }
}

// ---------------------------------------------------------------------------
// Kernel 2: causal flash attention, one wave per (qtile, head, batch).
// ---------------------------------------------------------------------------
__global__ __launch_bounds__(32) void attn_kernel(
    const __bf16* __restrict__ q, const __bf16* __restrict__ k,
    const __bf16* __restrict__ vT, __bf16* __restrict__ y)
{
  __shared__ __align__(64) __bf16 pls[512];   // P in packed A order
  const int lane = threadIdx.x;
  const int qt = blockIdx.x, h = blockIdx.y, b = blockIdx.z;
  const int t0 = qt * 16;
  const int nl = lane & 15;

  // Q fragments (A-layout): two 16B chunks each, aligned.
  const int trow = b * SEQ + min(t0 + nl, SEQ - 1);
  const __bf16* qbase = q + (size_t)trow * DMODEL + h * HDIM + ((lane >> 4) << 4);
  v16bf qa0 = cat8(*(const v8bf*)(qbase +  0), *(const v8bf*)(qbase + 8));
  v16bf qa1 = cat8(*(const v8bf*)(qbase + 32), *(const v8bf*)(qbase + 40));

  float mrow[8], lrow[8];
  v8f o0 = {}, o1 = {}, o2 = {}, o3 = {};
#pragma unroll
  for (int i = 0; i < 8; ++i) { mrow[i] = -__builtin_inff(); lrow[i] = 0.f; }

  const int tmax = min(t0 + 15, SEQ - 1);
  for (int k0 = 0; k0 <= tmax; k0 += 32) {
    // K^T B-fragments: contiguous 32B per lane (rows of k are 128B apart).
    const int kd0 = (lane >> 4) << 4;
    const __bf16* kp0 = k + (size_t)(b * SEQ + min(k0 + nl,      SEQ - 1)) * HDIM + kd0;
    const __bf16* kp1 = k + (size_t)(b * SEQ + min(k0 + 16 + nl, SEQ - 1)) * HDIM + kd0;
    v16bf kb00 = *(const v16bf*)(kp0);
    v16bf kb01 = *(const v16bf*)(kp0 + 32);
    v16bf kb10 = *(const v16bf*)(kp1);
    v16bf kb11 = *(const v16bf*)(kp1 + 32);

    v8f s0 = {}, s1 = {};
    s0 = wmma_bf16(qa0, kb00, s0);
    s0 = wmma_bf16(qa1, kb01, s0);
    s1 = wmma_bf16(qa0, kb10, s1);
    s1 = wmma_bf16(qa1, kb11, s1);

    // scale, causal mask, online softmax (lane = key column, element = row).
    float p0[8], p1[8], alpha[8];
#pragma unroll
    for (int i = 0; i < 8; ++i) {
      int mm = i + ((lane >> 4) << 3);
      int tq = t0 + mm;
      float a0 = s0[i] * 0.125f;
      float a1 = s1[i] * 0.125f;
      if (k0 + nl      > tq) a0 = -__builtin_inff();
      if (k0 + 16 + nl > tq) a1 = -__builtin_inff();
      float rm = fmaxf(a0, a1);
      for (int off = 1; off < 16; off <<= 1) rm = fmaxf(rm, __shfl_xor(rm, off, 32));
      float mnew = fmaxf(mrow[i], rm);
      alpha[i] = __expf(mrow[i] - mnew);
      mrow[i]  = mnew;
      p0[i] = __expf(a0 - mnew);
      p1[i] = __expf(a1 - mnew);
      float rs = p0[i] + p1[i];
      for (int off = 1; off < 16; off <<= 1) rs += __shfl_xor(rs, off, 32);
      lrow[i] = lrow[i] * alpha[i] + rs;
    }
#pragma unroll
    for (int i = 0; i < 8; ++i) {
      o0[i] *= alpha[i]; o1[i] *= alpha[i]; o2[i] *= alpha[i]; o3[i] *= alpha[i];
    }

    // P: D-layout -> packed A-layout via LDS.
    __syncthreads();
    {
      int hh = (nl >> 3) & 1;
      int i0 = (nl & 7);
#pragma unroll
      for (int i = 0; i < 8; ++i) {
        int mm = i + ((lane >> 4) << 3);
        int lt = (mm + (hh << 4)) << 4;
        pls[lt + i0]     = (__bf16)p0[i];   // keys k0+nl    (klocal 0..15)
        pls[lt + i0 + 8] = (__bf16)p1[i];   // keys k0+16+nl (klocal 16..31)
      }
    }
    __syncthreads();
    v16bf pa = *(const v16bf*)(pls + (lane << 4));

    // O += P * V using transposed V: two 8-bf16 runs per lane, 16 tokens apart.
    const int kl0 = (lane >> 4) << 3;
    const size_t vcol = (size_t)b * SEQ + k0 + kl0;
#pragma unroll
    for (int j = 0; j < 4; ++j) {
      const __bf16* vp = vT + (size_t)(j * 16 + nl) * NTOKP + vcol;
      v16bf vb = cat8(*(const v8bf*)(vp), *(const v8bf*)(vp + 16));
      if      (j == 0) o0 = wmma_bf16(pa, vb, o0);
      else if (j == 1) o1 = wmma_bf16(pa, vb, o1);
      else if (j == 2) o2 = wmma_bf16(pa, vb, o2);
      else             o3 = wmma_bf16(pa, vb, o3);
    }
  }

  // normalize and store y(b, t, h*64+d) as bf16.
#pragma unroll
  for (int i = 0; i < 8; ++i) {
    int mm = i + ((lane >> 4) << 3);
    int tq = t0 + mm;
    if (tq < SEQ) {
      float inv = 1.0f / lrow[i];
      __bf16* yp = y + (size_t)(b * SEQ + tq) * DMODEL + h * HDIM + nl;
      yp[0]  = (__bf16)(o0[i] * inv);
      yp[16] = (__bf16)(o1[i] * inv);
      yp[32] = (__bf16)(o2[i] * inv);
      yp[48] = (__bf16)(o3[i] * inv);
    }
  }
}

// ---------------------------------------------------------------------------
// Kernel 3: output projection y(bf16) @ Wo + bo -> fp32 out.
// ---------------------------------------------------------------------------
__global__ __launch_bounds__(256) void outproj_kernel(
    const __bf16* __restrict__ y, const __bf16* __restrict__ Wop,
    const float* __restrict__ bo, float* __restrict__ out)
{
  __shared__ __align__(64) __bf16 ys[16 * DMODEL];
  const int tid = threadIdx.x;
  const int m0  = blockIdx.x * 16;

  for (int idx = tid; idx < 16 * DMODEL; idx += 256) {
    int m = idx >> 10, d = idx & (DMODEL - 1);
    ys[a_pack_index(m, d)] = y[(size_t)(m0 + m) * DMODEL + d];
  }
  __syncthreads();

  const int wave = tid >> 5;
  const int lane = tid & 31;
  for (int nt = wave; nt < 64; nt += 8) {
    const __bf16* wbase = Wop + ((size_t)nt * NKK * 32 + lane) * 16;
    v8f acc = {};
    for (int kk = 0; kk < NKK; ++kk) {
      v16bf a  = *(const v16bf*)(ys + ((kk << 5) + lane) * 16);
      v16bf bf = *(const v16bf*)(wbase + (size_t)kk * 512);
      acc = wmma_bf16(a, bf, acc);
    }
    int n = nt * 16 + (lane & 15);
    float bias = bo[n];
#pragma unroll
    for (int i = 0; i < 8; ++i) {
      int mm = i + ((lane >> 4) << 3);
      out[(size_t)(m0 + mm) * DMODEL + n] = acc[i] + bias;
    }
  }
}

extern "C" void kernel_launch(void* const* d_in, const int* in_sizes, int n_in,
                              void* d_out, int out_size, void* d_ws, size_t ws_size,
                              hipStream_t stream) {
  const float* x  = (const float*)d_in[0];
  const float* Wq = (const float*)d_in[1];
  const float* bq = (const float*)d_in[2];
  const float* Wk = (const float*)d_in[3];
  const float* bk = (const float*)d_in[4];
  const float* Wv = (const float*)d_in[5];
  const float* bv = (const float*)d_in[6];
  const float* Wo = (const float*)d_in[7];
  const float* bo = (const float*)d_in[8];
  float* out = (float*)d_out;

  char* ws = (char*)d_ws;
  size_t off = 0;
  auto take = [&](size_t bytes) { char* p = ws + off; off += (bytes + 255) & ~(size_t)255; return p; };

  __bf16* qbuf = (__bf16*)take((size_t)NTOK * DMODEL * 2);
  __bf16* kbuf = (__bf16*)take((size_t)NTOK * HDIM * 2);
  __bf16* vT   = (__bf16*)take((size_t)HDIM * NTOKP * 2);
  __bf16* ybuf = (__bf16*)take((size_t)NTOK * DMODEL * 2);
  __bf16* Wqp  = (__bf16*)take((size_t)DMODEL * DMODEL * 2);
  __bf16* Wkp  = (__bf16*)take((size_t)DMODEL * HDIM * 2);
  __bf16* Wvp  = (__bf16*)take((size_t)DMODEL * HDIM * 2);
  __bf16* Wop  = (__bf16*)take((size_t)DMODEL * DMODEL * 2);
  float*  ct   = (float*)take((size_t)SEQ * (HDIM / 2) * 4);
  float*  st   = (float*)take((size_t)SEQ * (HDIM / 2) * 4);

  rope_table_kernel<<<(SEQ * (HDIM / 2) + 255) / 256, 256, 0, stream>>>(ct, st);
  pack_w_kernel<<<(DMODEL * DMODEL + 255) / 256, 256, 0, stream>>>(Wq, Wqp, DMODEL);
  pack_w_kernel<<<(DMODEL * HDIM + 255) / 256, 256, 0, stream>>>(Wk, Wkp, HDIM);
  pack_w_kernel<<<(DMODEL * HDIM + 255) / 256, 256, 0, stream>>>(Wv, Wvp, HDIM);
  pack_w_kernel<<<(DMODEL * DMODEL + 255) / 256, 256, 0, stream>>>(Wo, Wop, DMODEL);

  qkv_rope_kernel<<<NTOK / 16, 256, 0, stream>>>(x, Wqp, bq, Wkp, bk, Wvp, bv,
                                                 ct, st, qbuf, kbuf, vT);
  attn_kernel<<<dim3((SEQ + 15) / 16, NHEAD, NB), 32, 0, stream>>>(qbuf, kbuf, vT, ybuf);
  outproj_kernel<<<NTOK / 16, 256, 0, stream>>>(ybuf, Wop, bo, out);
}
